// MultiHeadAttention_65661460021965
// MI455X (gfx1250) — matrix operand order repaired and checked by reading, not compiled, not run on hardware
//
#include <hip/hip_runtime.h>
#include <hip/hip_bf16.h>

typedef _Float16 h16;
typedef __attribute__((ext_vector_type(16))) _Float16 v16h;
typedef __attribute__((ext_vector_type(8)))  _Float16 v8h;
typedef __attribute__((ext_vector_type(8)))  float    v8f;
typedef __attribute__((ext_vector_type(4)))  float    v4f;
typedef unsigned int u32x4 __attribute__((ext_vector_type(4)));
typedef int          i32x4 __attribute__((ext_vector_type(4)));
typedef int          i32x8 __attribute__((ext_vector_type(8)));

#define B_    2
#define T_    2048
#define H_    1024
#define NH_   16
#define HD_   64
#define M_    (B_ * T_)          // 4096 rows
#define NEGI  (-1.0e9f)
#define SCALE 0.125f             // 1/sqrt(64)

static __device__ __forceinline__ v16h cat8(v8h lo, v8h hi) {
    return __builtin_shufflevector(lo, hi, 0,1,2,3,4,5,6,7,8,9,10,11,12,13,14,15);
}

static __device__ __forceinline__ v8f wmma_f16(v16h a, v16h b, v8f c) {
    // D = A(16x32 f16) * B(32x16 f16) + C(16x16 f32)
    return __builtin_amdgcn_wmma_f32_16x16x32_f16(
        /*neg_a=*/false, a, /*neg_b=*/false, b,
        /*c_mod=*/(short)0, c, /*reuse_a=*/false, /*reuse_b=*/false);
}

// ---------------------------------------------------------------------------
// Tensor Data Mover: 2-D f16 tile Global -> LDS with optional row padding.
// D# layout per CDNA5 ISA.
//   tile_w / tensor_w in f16 elements; tensor_w also used as dim0 stride.
//   pad_i: pad after 2^(pad_i+1) DWORDs ; pad_a: pad by (pad_a+1) DWORDs.
// ---------------------------------------------------------------------------
static __device__ __forceinline__ void tdm_load_2d_f16(
    unsigned lds_byte_off, const void* gaddr,
    unsigned tile_w, unsigned tile_h,
    unsigned tensor_w, unsigned tensor_h,
    unsigned pad_i, unsigned pad_a, bool pad_en)
{
    unsigned long long ga = (unsigned long long)gaddr;
    u32x4 g0;
    g0.x = 1u;                                            // count=1 (valid user D#)
    g0.y = lds_byte_off;                                  // lds_addr
    g0.z = (unsigned)(ga & 0xFFFFFFFFu);                  // global_addr[31:0]
    g0.w = (unsigned)((ga >> 32) & 0x1FFFFFFu) | (2u << 30);  // addr[56:32] | type=2

    i32x8 g1 = {};
    unsigned d0 = (1u << 16);                             // data_size = 2 bytes
    if (pad_en) d0 |= (1u << 20) | (pad_i << 22) | (pad_a << 25);
    g1[0] = (int)d0;
    g1[1] = (int)((tensor_w & 0xFFFFu) << 16);            // tensor_dim0[15:0]
    g1[2] = (int)(((tensor_w >> 16) & 0xFFFFu) | ((tensor_h & 0xFFFFu) << 16));
    g1[3] = (int)(((tensor_h >> 16) & 0xFFFFu) | ((tile_w & 0xFFFFu) << 16));
    g1[4] = (int)(tile_h & 0xFFFFu);                      // tile_dim1 (tile_dim2=0)
    g1[5] = (int)tensor_w;                                // tensor_dim0_stride[31:0]
    g1[6] = 0;
    g1[7] = 0;

    i32x4 g2 = {};
    i32x4 g3 = {};
#if __clang_major__ >= 23
    i32x8 g4 = {};
    __builtin_amdgcn_tensor_load_to_lds(g0, g1, g2, g3, g4, 0);
#else
    __builtin_amdgcn_tensor_load_to_lds(g0, g1, g2, g3, 0);
#endif
}

// ---------------------------------------------------------------------------
// GEMM 1: Y = X(f32)[M,K] @ W(f32)[N,K]^T + bias, written f16 into [B,NH,T,HD]
// Block: 256 threads (8 waves), tile 64(M) x 128(N), K-chunks of 32.
// (f32 inputs need VALU conversion, so staging stays on the vector path.)
// ---------------------------------------------------------------------------
__global__ __launch_bounds__(256)
void mha_gemm_qkv(const float* __restrict__ X, const float* __restrict__ W,
                  const float* __restrict__ bias, h16* __restrict__ dst) {
    __shared__ h16 Xs[64 * 40];    // pitch 40 halves (80B, 16B aligned)
    __shared__ h16 Ws[128 * 40];

    const int tid  = threadIdx.x;
    const int lane = tid & 31;
    const int wid  = tid >> 5;
    const int wm   = wid & 3;
    const int wn   = wid >> 2;
    const int ln   = lane & 15;
    const int lh   = lane >> 4;
    const int tileM = blockIdx.x * 64;
    const int tileN = blockIdx.y * 128;

    v8f acc[4] = {};

    for (int kc = 0; kc < H_; kc += 32) {
        __syncthreads();
        {   // stage X tile: 64x32 f32 -> f16
            const int r = tid >> 2, c = (tid & 3) * 8;
            const float* src = X + (size_t)(tileM + r) * H_ + kc + c;
            __builtin_prefetch(src + 32, 0, 0);
            v4f f0 = *(const v4f*)src;
            v4f f1 = *(const v4f*)(src + 4);
            h16* d = &Xs[r * 40 + c];
            d[0]=(h16)f0.x; d[1]=(h16)f0.y; d[2]=(h16)f0.z; d[3]=(h16)f0.w;
            d[4]=(h16)f1.x; d[5]=(h16)f1.y; d[6]=(h16)f1.z; d[7]=(h16)f1.w;
        }
        {   // stage W tile: 128x32 f32 -> f16
            const int r = tid >> 1, c = (tid & 1) * 16;
            const float* src = W + (size_t)(tileN + r) * H_ + kc + c;
            v4f f0 = *(const v4f*)src;
            v4f f1 = *(const v4f*)(src + 4);
            v4f f2 = *(const v4f*)(src + 8);
            v4f f3 = *(const v4f*)(src + 12);
            h16* d = &Ws[r * 40 + c];
            d[0]=(h16)f0.x;  d[1]=(h16)f0.y;  d[2]=(h16)f0.z;  d[3]=(h16)f0.w;
            d[4]=(h16)f1.x;  d[5]=(h16)f1.y;  d[6]=(h16)f1.z;  d[7]=(h16)f1.w;
            d[8]=(h16)f2.x;  d[9]=(h16)f2.y;  d[10]=(h16)f2.z; d[11]=(h16)f2.w;
            d[12]=(h16)f3.x; d[13]=(h16)f3.y; d[14]=(h16)f3.z; d[15]=(h16)f3.w;
        }
        __syncthreads();

        const h16* ab = &Xs[(wm * 16 + ln) * 40 + lh * 8];
        v16h a = cat8(*(const v8h*)ab, *(const v8h*)(ab + 16));
        #pragma unroll
        for (int t = 0; t < 4; ++t) {
            const h16* bb = &Ws[(wn * 64 + t * 16 + ln) * 40 + lh * 8];
            v16h b = cat8(*(const v8h*)bb, *(const v8h*)(bb + 16));
            acc[t] = wmma_f16(a, b, acc[t]);
        }
    }

    #pragma unroll
    for (int t = 0; t < 4; ++t) {
        const int col = tileN + wn * 64 + t * 16 + ln;
        const float bv = bias[col];
        const int nh = col >> 6, hd = col & 63;
        #pragma unroll
        for (int r = 0; r < 8; ++r) {
            const int row = tileM + wm * 16 + lh * 8 + r;
            const int bb = row >> 11, tt = row & (T_ - 1);
            dst[((size_t)(bb * NH_ + nh) * T_ + tt) * HD_ + hd] = (h16)(acc[t][r] + bv);
        }
    }
}

// ---------------------------------------------------------------------------
// GEMM 2: out = Oh(f16)[M,H] @ Wo(f32)[H,H]^T + bias, f32 out.
// X tile staged by TDM (f16, padded pitch 40); W staged with conversion.
// Dynamic LDS only => descriptor lds_addr offsets start at 0.
// ---------------------------------------------------------------------------
#define GO_XS_OFF   0                       // 64*40*2 = 5120 B
#define GO_WS_OFF   5120                    // 128*40*2 = 10240 B
#define GO_LDS_SZ   15360

__global__ __launch_bounds__(256)
void mha_gemm_out(const h16* __restrict__ Xh, const float* __restrict__ W,
                  const float* __restrict__ bias, float* __restrict__ out) {
    extern __shared__ __align__(16) char smem[];
    h16* Xs = (h16*)(smem + GO_XS_OFF);
    h16* Ws = (h16*)(smem + GO_WS_OFF);

    const int tid  = threadIdx.x;
    const int lane = tid & 31;
    const int wid  = tid >> 5;
    const int wm   = wid & 3;
    const int wn   = wid >> 2;
    const int ln   = lane & 15;
    const int lh   = lane >> 4;
    const int tileM = blockIdx.x * 64;
    const int tileN = blockIdx.y * 128;

    v8f acc[4] = {};

    for (int kc = 0; kc < H_; kc += 32) {
        __syncthreads();
        if (wid == 0) {
            // TDM: 64 rows x 32 halves from Xh, pad 16 DWORDs -> +4 DWORDs (pitch 40)
            tdm_load_2d_f16(GO_XS_OFF, Xh + (size_t)tileM * H_ + kc,
                            /*tile_w=*/32, /*tile_h=*/64,
                            /*tensor_w=*/H_, /*tensor_h=*/M_ - tileM,
                            /*pad_i=*/3, /*pad_a=*/3, true);
        }
        {   // stage W tile: 128x32 f32 -> f16 (VALU convert path)
            const int r = tid >> 1, c = (tid & 1) * 16;
            const float* src = W + (size_t)(tileN + r) * H_ + kc + c;
            v4f f0 = *(const v4f*)src;
            v4f f1 = *(const v4f*)(src + 4);
            v4f f2 = *(const v4f*)(src + 8);
            v4f f3 = *(const v4f*)(src + 12);
            h16* d = &Ws[r * 40 + c];
            d[0]=(h16)f0.x;  d[1]=(h16)f0.y;  d[2]=(h16)f0.z;  d[3]=(h16)f0.w;
            d[4]=(h16)f1.x;  d[5]=(h16)f1.y;  d[6]=(h16)f1.z;  d[7]=(h16)f1.w;
            d[8]=(h16)f2.x;  d[9]=(h16)f2.y;  d[10]=(h16)f2.z; d[11]=(h16)f2.w;
            d[12]=(h16)f3.x; d[13]=(h16)f3.y; d[14]=(h16)f3.z; d[15]=(h16)f3.w;
        }
        if (wid == 0) __builtin_amdgcn_s_wait_tensorcnt(0);
        __syncthreads();

        const h16* ab = &Xs[(wm * 16 + ln) * 40 + lh * 8];
        v16h a = cat8(*(const v8h*)ab, *(const v8h*)(ab + 16));
        #pragma unroll
        for (int t = 0; t < 4; ++t) {
            const h16* bb = &Ws[(wn * 64 + t * 16 + ln) * 40 + lh * 8];
            v16h b = cat8(*(const v8h*)bb, *(const v8h*)(bb + 16));
            acc[t] = wmma_f16(a, b, acc[t]);
        }
    }

    #pragma unroll
    for (int t = 0; t < 4; ++t) {
        const int col = tileN + wn * 64 + t * 16 + ln;
        const float bv = bias[col];
        #pragma unroll
        for (int r = 0; r < 8; ++r) {
            const int row = tileM + wm * 16 + lh * 8 + r;
            out[(size_t)row * H_ + col] = acc[t][r] + bv;
        }
    }
}

// ---------------------------------------------------------------------------
// Flash-style causal attention. Grid (B*NH, T/64); block 128 (4 waves).
// Q and K tiles staged by TDM with row padding (pitch 72); V transposed by VALU.
// ---------------------------------------------------------------------------
#define AT_QS_OFF   0                       // 64*72*2 = 9216 B
#define AT_KS_OFF   9216                    // 32*72*2 = 4608 B
#define AT_VT_OFF   13824                   // 64*40*2 = 5120 B
#define AT_PS_OFF   18944                   // 4*16*40*2 = 5120 B
#define AT_LDS_SZ   24064

__global__ __launch_bounds__(128)
void mha_attn(const h16* __restrict__ Qh, const h16* __restrict__ Kh,
              const h16* __restrict__ Vh, h16* __restrict__ Oh) {
    extern __shared__ __align__(16) char smem[];
    h16* Qs = (h16*)(smem + AT_QS_OFF);     // [64][72]
    h16* Ks = (h16*)(smem + AT_KS_OFF);     // [32][72]
    h16* Vt = (h16*)(smem + AT_VT_OFF);     // [64][40] transposed
    h16* Ps = (h16*)(smem + AT_PS_OFF);     // [4][16][40]

    const int tid  = threadIdx.x;
    const int lane = tid & 31;
    const int wid  = tid >> 5;
    const int ln   = lane & 15;
    const int lh   = lane >> 4;

    const int bh    = blockIdx.x;
    const int qbase = blockIdx.y * 64;
    const int b     = bh >> 4;
    const int nh    = bh & 15;

    const h16* Qp = Qh + (size_t)bh * T_ * HD_;
    const h16* Kp = Kh + (size_t)bh * T_ * HD_;
    const h16* Vp = Vh + (size_t)bh * T_ * HD_;

    if (wid == 0) {
        // TDM: Q tile 64x64 halves, pad 32 DWORDs -> +4 DWORDs (pitch 72)
        tdm_load_2d_f16(AT_QS_OFF, Qp + (size_t)qbase * HD_,
                        /*tile_w=*/HD_, /*tile_h=*/64,
                        /*tensor_w=*/HD_, /*tensor_h=*/T_ - qbase,
                        /*pad_i=*/4, /*pad_a=*/3, true);
        __builtin_amdgcn_s_wait_tensorcnt(0);
    }

    v8f acc[4] = {};
    float mrow[8], lrow[8];
    #pragma unroll
    for (int r = 0; r < 8; ++r) { mrow[r] = NEGI; lrow[r] = 0.0f; }

    const int kv_end = qbase + 64;
    for (int kv = 0; kv < kv_end; kv += 32) {
        __syncthreads();
        if (wid == 0) {
            // TDM: K tile 32x64 halves, pitch 72
            tdm_load_2d_f16(AT_KS_OFF, Kp + (size_t)kv * HD_,
                            /*tile_w=*/HD_, /*tile_h=*/32,
                            /*tensor_w=*/HD_, /*tensor_h=*/T_ - kv,
                            /*pad_i=*/4, /*pad_a=*/3, true);
        }
        {   // stage V tile transposed into Vt[d][k] (VALU path: TDM cannot transpose)
            const int r = tid >> 2, c = (tid & 3) * 16;
            const h16* vsrc = Vp + (size_t)(kv + r) * HD_ + c;
            v8h v0 = *(const v8h*)(vsrc);
            v8h v1 = *(const v8h*)(vsrc + 8);
            #pragma unroll
            for (int i = 0; i < 8; ++i) {
                Vt[(c + i) * 40 + r]     = v0[i];
                Vt[(c + 8 + i) * 40 + r] = v1[i];
            }
        }
        if (wid == 0) __builtin_amdgcn_s_wait_tensorcnt(0);
        __syncthreads();

        // Q A-fragments for d-chunks 0 and 1
        const h16* qb = &Qs[(wid * 16 + ln) * 72];
        v16h a0 = cat8(*(const v8h*)(qb + lh * 8),      *(const v8h*)(qb + 16 + lh * 8));
        v16h a1 = cat8(*(const v8h*)(qb + 32 + lh * 8), *(const v8h*)(qb + 48 + lh * 8));

        // S = Q K^T for two 16-key subtiles
        v8f S[2];
        #pragma unroll
        for (int j = 0; j < 2; ++j) {
            const h16* kb = &Ks[(j * 16 + ln) * 72];
            v16h b0 = cat8(*(const v8h*)(kb + lh * 8),      *(const v8h*)(kb + 16 + lh * 8));
            v16h b1 = cat8(*(const v8h*)(kb + 32 + lh * 8), *(const v8h*)(kb + 48 + lh * 8));
            v8f z = {};
            v8f s = wmma_f16(a0, b0, z);
            s = wmma_f16(a1, b1, s);
            const int colg = kv + j * 16 + ln;
            #pragma unroll
            for (int r = 0; r < 8; ++r) {
                const int rowg = qbase + wid * 16 + lh * 8 + r;
                float sv = s[r] * SCALE;
                S[j][r] = (colg > rowg) ? NEGI : sv;   // causal mask
            }
        }

        // online softmax: per-row reductions within each 16-lane half
        #pragma unroll
        for (int r = 0; r < 8; ++r) {
            float mx = fmaxf(S[0][r], S[1][r]);
            #pragma unroll
            for (int sh = 1; sh < 16; sh <<= 1)
                mx = fmaxf(mx, __shfl_xor(mx, sh, 32));
            const float mnew  = fmaxf(mrow[r], mx);
            const float alpha = __expf(mrow[r] - mnew);
            const float p0 = __expf(S[0][r] - mnew);
            const float p1 = __expf(S[1][r] - mnew);
            S[0][r] = p0; S[1][r] = p1;
            float rs = p0 + p1;
            #pragma unroll
            for (int sh = 1; sh < 16; sh <<= 1)
                rs += __shfl_xor(rs, sh, 32);
            lrow[r] = lrow[r] * alpha + rs;
            mrow[r] = mnew;
            #pragma unroll
            for (int t = 0; t < 4; ++t) acc[t][r] *= alpha;
        }

        // stage P (C-layout -> row-major 16x32) for A-fragment reload
        h16* pb = &Ps[wid * 16 * 40];
        #pragma unroll
        for (int j = 0; j < 2; ++j)
            #pragma unroll
            for (int r = 0; r < 8; ++r)
                pb[(lh * 8 + r) * 40 + j * 16 + ln] = (h16)S[j][r];
        __syncthreads();

        const h16* pf = &Ps[wid * 16 * 40 + ln * 40];
        v16h ap = cat8(*(const v8h*)(pf + lh * 8), *(const v8h*)(pf + 16 + lh * 8));
        #pragma unroll
        for (int t = 0; t < 4; ++t) {
            const h16* vb = &Vt[(t * 16 + ln) * 40 + lh * 8];
            v16h bv = cat8(*(const v8h*)vb, *(const v8h*)(vb + 16));
            acc[t] = wmma_f16(ap, bv, acc[t]);
        }
    }

    // epilogue: normalize and write Oh as [B, T, H] f16
    #pragma unroll
    for (int t = 0; t < 4; ++t) {
        const int col = nh * HD_ + t * 16 + ln;
        #pragma unroll
        for (int r = 0; r < 8; ++r) {
            const int row = qbase + wid * 16 + lh * 8 + r;
            Oh[((size_t)b * T_ + row) * H_ + col] = (h16)(acc[t][r] / lrow[r]);
        }
    }
}

// ---------------------------------------------------------------------------
extern "C" void kernel_launch(void* const* d_in, const int* in_sizes, int n_in,
                              void* d_out, int out_size, void* d_ws, size_t ws_size,
                              hipStream_t stream) {
    const float* q   = (const float*)d_in[0];
    const float* k   = (const float*)d_in[1];
    const float* v   = (const float*)d_in[2];
    // d_in[3] = causal mask, implicit in kernel
    const float* w_q = (const float*)d_in[4];
    const float* b_q = (const float*)d_in[5];
    const float* w_k = (const float*)d_in[6];
    const float* b_k = (const float*)d_in[7];
    const float* w_v = (const float*)d_in[8];
    const float* b_v = (const float*)d_in[9];
    const float* w_o = (const float*)d_in[10];
    const float* b_o = (const float*)d_in[11];
    float* out = (float*)d_out;

    const size_t elems = (size_t)B_ * NH_ * T_ * HD_;   // 4M f16 each
    h16* Qh = (h16*)d_ws;
    h16* Kh = Qh + elems;
    h16* Vh = Kh + elems;
    h16* Oh = Vh + elems;

    dim3 ggrid(M_ / 64, H_ / 128);
    mha_gemm_qkv<<<ggrid, 256, 0, stream>>>(q, w_q, b_q, Qh);
    mha_gemm_qkv<<<ggrid, 256, 0, stream>>>(k, w_k, b_k, Kh);
    mha_gemm_qkv<<<ggrid, 256, 0, stream>>>(v, w_v, b_v, Vh);

    mha_attn<<<dim3(B_ * NH_, T_ / 64), 128, AT_LDS_SZ, stream>>>(Qh, Kh, Vh, Oh);

    mha_gemm_out<<<ggrid, 256, GO_LDS_SZ, stream>>>(Oh, w_o, b_o, out);
}
